// SpeakerMemory_18150531792939
// MI455X (gfx1250) — compile-verified, matrix operand order checked by reading, not run on hardware
//
#include <hip/hip_runtime.h>

// ---------------------------------------------------------------------------
// Speaker-memory GRU for MI455X (gfx1250, wave32, WMMA).
//
//   Kernel 1: convert W_ih / W_hh (f32, [768][256] row-major) -> bf16 image
//             in d_ws (786 KB, stays hot in the 192 MB L2).
//   Kernel 2: persistent scan. 8 workgroups x 16 batch rows (batch rows are
//             fully independent -> zero inter-WG sync).
//     - LDS: speaker bank mem[16][10][256] f32 (160 KB; needs CDNA5's
//       320 KB WGP LDS) + x slice + gate accumulators.
//     - per step: dual GEMM [16,256]x[256,768] via v_wmma_f32_16x16x32_bf16.
//       All 16 A-fragments built once per step and held in registers (reused
//       by all 6 n-tiles); per n-tile an 8-deep in-place WMMA accumulation
//       chain with only 2 live accumulators, stored to LDS immediately.
//       B = bf16 weights streamed from L2 each step as global_load_b128 with
//       folded 24-bit immediates; an opaque per-iteration byte offset on the
//       SGPR base (asm "+s" on an integer, preserving addrspace provenance)
//       blocks LICM from hoisting the weight stream (-> no spills).
//     - x slice staged global->LDS with global_load_async_to_lds_b128
//       (ASYNCcnt path), completed by s_wait_asynccnt + barrier.
//     - gate math on v_exp_f32 + v_rcp_f32 (1-ulp rcp instead of IEEE
//       divide chains: gates saturate, accuracy is ample).
// ---------------------------------------------------------------------------

typedef __attribute__((ext_vector_type(16))) __bf16 v16bf;
typedef __attribute__((ext_vector_type(8)))  __bf16 v8bf;
typedef __attribute__((ext_vector_type(8)))  float  v8f;
typedef __attribute__((ext_vector_type(4)))  float  v4f;

#define T_STEPS 1024
#define DIM     256      // D_IN == D_MEM == 256
#define NG      768      // 3 * H
#define SPK     10       // SPEAKERS_MAX
#define BW      16       // batch rows per workgroup
#define NWG     8        // 128 / BW
#define NT_PW   6        // n-tiles (16 cols) per wave: 48 tiles / 8 waves

// Build an A fragment (16-bit A 16x32 layout): this lane holds row m = lane&15,
// K offsets koff+[0..8) and koff+16+[0..8) within the 32-wide K chunk.
// Source is f32 in LDS; convert to bf16 (RNE) while packing.
__device__ __forceinline__ v16bf make_a_frag(const float* __restrict__ p) {
  v4f a0 = *(const v4f*)(p + 0);
  v4f a1 = *(const v4f*)(p + 4);
  v4f b0 = *(const v4f*)(p + 16);
  v4f b1 = *(const v4f*)(p + 20);
  v16bf r;
#pragma unroll
  for (int i = 0; i < 4; ++i) {
    r[i]      = (__bf16)a0[i];
    r[4 + i]  = (__bf16)a1[i];
    r[8 + i]  = (__bf16)b0[i];
    r[12 + i] = (__bf16)b1[i];
  }
  return r;
}

// B fragment (K x 16): lane holds W row n = n0 + (lane&15) (column of W^T),
// same two-chunk K pattern. Weights already bf16: two 16-byte loads.
__device__ __forceinline__ v16bf load_b_frag(const __bf16* __restrict__ p) {
  v8bf lo = *(const v8bf*)(p + 0);
  v8bf hi = *(const v8bf*)(p + 16);
  return __builtin_shufflevector(lo, hi, 0, 1, 2, 3, 4, 5, 6, 7,
                                 8, 9, 10, 11, 12, 13, 14, 15);
}

// fast transcendentals: v_exp_f32 + single-instruction v_rcp_f32 (~1 ulp),
// plenty for saturating GRU gates.
__device__ __forceinline__ float sigmoid_fast(float x) {
  return __builtin_amdgcn_rcpf(1.f + __expf(-x));
}
__device__ __forceinline__ float tanh_fast(float x) {
  // tanh(x) = 1 - 2/(e^{2x}+1)
  return 1.f - 2.f * __builtin_amdgcn_rcpf(__expf(2.f * x) + 1.f);
}

__global__ void SpeakerMemory_convert_w(const float* __restrict__ wih,
                                        const float* __restrict__ whh,
                                        __bf16* __restrict__ obf) {
  int i = blockIdx.x * 256 + threadIdx.x;
  const int N = NG * DIM;
  if (i < N) {
    obf[i] = (__bf16)wih[i];
  } else if (i < 2 * N) {
    obf[i] = (__bf16)whh[i - N];
  }
}

__global__ __launch_bounds__(256)
void SpeakerMemory_scan(const float* __restrict__ x_in,
                        const int*   __restrict__ speakers,
                        const __bf16* __restrict__ wih_bf,   // [768][256] bf16
                        const __bf16* __restrict__ whh_bf,   // [768][256] bf16
                        const float* __restrict__ b_ih,
                        const float* __restrict__ b_hh,
                        float* __restrict__ out) {
  extern __shared__ float smem[];
  float* memb = smem;                      // [16][10][256]  160 KB speaker bank
  float* xbuf = memb + BW * SPK * DIM;     // [16][256]      x slice (f32)
  float* accx = xbuf + BW * DIM;           // [16][768]      x-gate accumulators
  float* acch = accx + BW * NG;            // [16][768]      h-gate accumulators
  int*   srow = (int*)(acch + BW * NG);    // [16]           speaker id per row

  const int tid  = threadIdx.x;
  const int lane = tid & 31;
  const int wv   = tid >> 5;               // wave id, 0..7
  const int b0   = blockIdx.x * BW;        // first batch row of this WG

  // zero the speaker memory bank (reference mem0 = zeros)
  for (int i = tid; i < BW * SPK * DIM; i += 256) memb[i] = 0.f;

  // hoist gate biases (t-invariant), C/D layout: N = lane&15
  float bx[NT_PW], bh[NT_PW];
#pragma unroll
  for (int i = 0; i < NT_PW; ++i) {
    int col = (wv * NT_PW + i) * 16 + (lane & 15);
    bx[i] = b_ih[col];
    bh[i] = b_hh[col];
  }

  const int m    = lane & 15;              // A-row / B-col within a tile
  const int koff = (lane >> 4) * 8;        // K sub-chunk select per lane half

  // Single hoisted 32-bit divergent element offset for the weight streams;
  // everything else folds into the 24-bit instruction immediate.
  const unsigned abase =
      (unsigned)((wv * NT_PW * 16 + m) * DIM + koff);

  // async-staging addresses for this thread: 16 contiguous floats of one row
  const int xr_row = tid >> 4;             // 0..15
  const int xr_col = (tid & 15) * 16;      // 0..240
  // LDS byte offset (flat LDS address: low 32 bits are the LDS offset)
  const unsigned lds_x = (unsigned)(uintptr_t)(xbuf + xr_row * DIM + xr_col);
  const float* gx_base =
      x_in + ((size_t)(b0 + xr_row) * T_STEPS) * DIM + xr_col;

  __syncthreads();

  for (int t = 0; t < T_STEPS; ++t) {
    // Opaque zero byte-offset on the UNIFORM base, re-defined every
    // iteration: derived load addresses are not loop-invariant (no weight
    // hoisting / spilling), pointer provenance stays global.
    unsigned woff = 0;
    asm volatile("" : "+s"(woff));
    const __bf16* wih_p = wih_bf + woff + abase;
    const __bf16* whh_p = whh_bf + woff + abase;

    // -------- phase 1: async-stage x_in[:, t, :] slice + speaker ids -----
    {
      const float* gsrc = gx_base + (size_t)t * DIM;
#pragma unroll
      for (int q = 0; q < 4; ++q) {
        asm volatile("global_load_async_to_lds_b128 %0, %1, off"
                     :
                     : "v"(lds_x + 16u * q), "v"(gsrc + 4 * q)
                     : "memory");
      }
      if (tid < BW) {
        int s = speakers[(size_t)(b0 + tid) * T_STEPS + t];
        srow[tid] = s < 0 ? 0 : (s > SPK - 1 ? SPK - 1 : s);
      }
      asm volatile("s_wait_asynccnt 0x0" ::: "memory");
    }
    __syncthreads();

    // -------- phase 2: dual GEMM (xg = x W_ih^T + b_ih ; hg = h W_hh^T + b_hh)
    const int sm = srow[m];
    const float* hrow = memb + (m * SPK + sm) * DIM;   // gathered hidden state
    const float* xrow = xbuf + m * DIM;

    // Build all A fragments once; register-resident across the n-tile loop.
    v16bf axf[8], ahf[8];
#pragma unroll
    for (int kc = 0; kc < 8; ++kc) {
      const int k0 = kc * 32 + koff;
      axf[kc] = make_a_frag(xrow + k0);
      ahf[kc] = make_a_frag(hrow + k0);
    }

#pragma unroll
    for (int i = 0; i < NT_PW; ++i) {
      v8f ax, ah;                          // only 2 live accumulators
#pragma unroll
      for (int e = 0; e < 8; ++e) { ax[e] = bx[i]; ah[e] = bh[i]; }
#pragma unroll
      for (int kc = 0; kc < 8; ++kc) {
        // constant element offset per (i,kc): folds into the load immediate
        const unsigned coff = (unsigned)(i * 16 * DIM + kc * 32);
        v16bf bif = load_b_frag(wih_p + coff);
        v16bf bhf = load_b_frag(whh_p + coff);
        ax = __builtin_amdgcn_wmma_f32_16x16x32_bf16(
            false, axf[kc], false, bif, (short)0, ax, false, false);
        ah = __builtin_amdgcn_wmma_f32_16x16x32_bf16(
            false, ahf[kc], false, bhf, (short)0, ah, false, false);
      }
      // store this tile's accumulators right away (short acc lifetime)
      const int col = (wv * NT_PW + i) * 16 + m;
#pragma unroll
      for (int v = 0; v < 8; ++v) {
        const int row = v + 8 * (lane >> 4);
        accx[row * NG + col] = ax[v];
        acch[row * NG + col] = ah[v];
      }
    }
    __syncthreads();

    // -------- phase 3: GRU gates + speaker-bank update + output ----------
    {
      const int r  = tid >> 4;
      const int j0 = (tid & 15) * 16;
      const int s  = srow[r];
      float* hbank = memb + (r * SPK + s) * DIM + j0;
      float* orow  = out + ((size_t)(b0 + r) * T_STEPS + t) * DIM + j0;
      const float* ax = accx + r * NG + j0;
      const float* ah = acch + r * NG + j0;
#pragma unroll
      for (int q = 0; q < 4; ++q) {
        v4f xr = *(const v4f*)(ax + 4 * q);
        v4f xz = *(const v4f*)(ax + 256 + 4 * q);
        v4f xn = *(const v4f*)(ax + 512 + 4 * q);
        v4f hr = *(const v4f*)(ah + 4 * q);
        v4f hz = *(const v4f*)(ah + 256 + 4 * q);
        v4f hn = *(const v4f*)(ah + 512 + 4 * q);
        v4f ho = *(const v4f*)(hbank + 4 * q);
        v4f hnew;
#pragma unroll
        for (int e = 0; e < 4; ++e) {
          const float rg = sigmoid_fast(xr[e] + hr[e]);
          const float zg = sigmoid_fast(xz[e] + hz[e]);
          const float ng = tanh_fast(xn[e] + rg * hn[e]);
          // (1-z)*n + z*h  ==  (1-z)*(n-h) + h
          hnew[e] = (1.f - zg) * (ng - ho[e]) + ho[e];
        }
        *(v4f*)(hbank + 4 * q) = hnew;
        *(v4f*)(orow + 4 * q) = hnew;
      }
    }
    __syncthreads();
  }
}

extern "C" void kernel_launch(void* const* d_in, const int* in_sizes, int n_in,
                              void* d_out, int out_size, void* d_ws, size_t ws_size,
                              hipStream_t stream) {
  const float* x_in = (const float*)d_in[0];
  const int*   spk  = (const int*)d_in[1];
  const float* wih  = (const float*)d_in[2];
  const float* whh  = (const float*)d_in[3];
  const float* bih  = (const float*)d_in[4];
  const float* bhh  = (const float*)d_in[5];
  float* out = (float*)d_out;

  __bf16* wbf = (__bf16*)d_ws;                 // 2 * 768*256 bf16 = 786 KB

  const int nW = NG * DIM;
  SpeakerMemory_convert_w<<<(2 * nW + 255) / 256, 256, 0, stream>>>(wih, whh, wbf);

  const size_t smem_bytes =
      (size_t)(BW * SPK * DIM + BW * DIM + 2 * BW * NG) * sizeof(float) +
      BW * sizeof(int);                        // ~272 KB, fits 320 KB WGP LDS
  SpeakerMemory_scan<<<NWG, 256, smem_bytes, stream>>>(
      x_in, spk, wbf, wbf + nW, bih, bhh, out);
}
